// RelativeGlobalAttention_80977313399035
// MI455X (gfx1250) — matrix operand order repaired and checked by reading, not compiled, not run on hardware
//
#include <hip/hip_runtime.h>

typedef __attribute__((ext_vector_type(16))) __bf16 v16bf;
typedef __attribute__((ext_vector_type(8)))  __bf16 v8bf;
typedef __attribute__((ext_vector_type(8)))  float  v8f;

#define B_   4
#define H_   4
#define L_   2048
#define D_   256
#define DQK_ 32
#define DH_  64
#define EPAD_ (L_ + 128)      // E padded with zero rows for the skew band
#define QESTR 97
#define PSTR  72              // 72*2 = 144 bytes, multiple of 16 -> aligned chunks

// ---------- helpers ----------------------------------------------------------
__device__ __forceinline__ __bf16 f2bf(float f) {
  unsigned u = __builtin_bit_cast(unsigned, f);
  unsigned r = u + 0x7FFFu + ((u >> 16) & 1u);       // round-to-nearest-even
  unsigned short h = (unsigned short)(r >> 16);
  return __builtin_bit_cast(__bf16, h);
}

// B-matrix (32x16 bf16) fragment: lane's 16 values are contiguous at k=16*half
__device__ __forceinline__ v16bf load_b_frag(const __bf16* row, int half) {
  return *(const v16bf*)(row + 16 * half);
}
// A-matrix (16x32 bf16) fragment: two contiguous 8-element chunks
// k in [16h, 16h+8) then [16h+8, 16h+16)
__device__ __forceinline__ v16bf load_a_frag(const __bf16* row, int half) {
  v8bf lo = *(const v8bf*)(row + 16 * half);
  v8bf hi = *(const v8bf*)(row + 16 * half + 8);
  return __builtin_shufflevector(lo, hi, 0, 1, 2, 3, 4, 5, 6, 7,
                                 8, 9, 10, 11, 12, 13, 14, 15);
}

// gfx1250 async global->LDS copy (16B per lane), tracked by ASYNCcnt
__device__ __forceinline__ void async_copy_b128(unsigned ldsOff, const void* g) {
  asm volatile("global_load_async_to_lds_b128 %0, %1, off"
               :: "v"(ldsOff), "v"(g) : "memory");
}
__device__ __forceinline__ void wait_async0() {
  asm volatile("s_wait_asynccnt 0" ::: "memory");
}

// ---------- fp32 -> bf16 conversion ------------------------------------------
__global__ void cvt_f32_bf16(const float* __restrict__ src, __bf16* __restrict__ dst, int n) {
  int i = blockIdx.x * blockDim.x + threadIdx.x;
  if (i < n) dst[i] = f2bf(src[i]);
}
// src[K,N] fp32 -> dst[N,K] bf16 (weight transpose so GEMM B-frags are contiguous)
__global__ void cvt_f32_bf16_T(const float* __restrict__ src, __bf16* __restrict__ dst,
                               int K, int N) {
  int i = blockIdx.x * blockDim.x + threadIdx.x;
  if (i < K * N) {
    int k = i / N, n = i % N;
    dst[(size_t)n * K + k] = f2bf(src[i]);
  }
}
__global__ void zero_bf16(__bf16* __restrict__ dst, int n) {
  int i = blockIdx.x * blockDim.x + threadIdx.x;
  if (i < n) dst[i] = f2bf(0.0f);
}

// ---------- generic 64x64-tile bf16 WMMA GEMM (K = 256) ----------------------
// C[m,n] = sum_k A[m,k] * Wt[n,k] + bias[n]     (Wt is pre-transposed [N,K])
// Software-pipelined: fragments for k+32 are loaded while WMMAs for k execute.
__global__ __launch_bounds__(128) void gemm_bf16(
    const __bf16* __restrict__ A, const __bf16* __restrict__ Wt,
    const float* __restrict__ bias,
    __bf16* __restrict__ outSplit, float* __restrict__ outPlain,
    int N, int headDim, int vTrans)
{
  const int wave = threadIdx.x >> 5;
  const int lane = threadIdx.x & 31;
  const int ln   = lane & 15;
  const int half = lane >> 4;
  const int row0 = blockIdx.x * 64 + wave * 16;
  const int col0 = blockIdx.y * 64;
  const int K    = D_;

  v8f acc[4] = {};
  const __bf16* Arow  = A  + (size_t)(row0 + ln) * K;
  const __bf16* Wrow0 = Wt + (size_t)(col0 + ln) * K;

  v16bf af = load_a_frag(Arow, half);
  v16bf wf[4];
#pragma unroll
  for (int nt = 0; nt < 4; ++nt)
    wf[nt] = load_b_frag(Wrow0 + (size_t)nt * 16 * K, half);

#pragma unroll 2
  for (int kk = 0; kk < K; kk += 32) {
    v16bf afn = {};
    v16bf wfn[4] = {};
    if (kk + 32 < K) {
      afn = load_a_frag(Arow + kk + 32, half);
#pragma unroll
      for (int nt = 0; nt < 4; ++nt)
        wfn[nt] = load_b_frag(Wrow0 + (size_t)nt * 16 * K + kk + 32, half);
    }
#pragma unroll
    for (int nt = 0; nt < 4; ++nt)
      acc[nt] = __builtin_amdgcn_wmma_f32_16x16x32_bf16(
          false, af, false, wf[nt], (short)0, acc[nt], false, false);
    af = afn;
#pragma unroll
    for (int nt = 0; nt < 4; ++nt) wf[nt] = wfn[nt];
  }

#pragma unroll
  for (int nt = 0; nt < 4; ++nt)
#pragma unroll
    for (int v = 0; v < 8; ++v) {
      int m = row0 + v + 8 * half;
      int n = col0 + nt * 16 + ln;
      float val = acc[nt][v] + bias[n];
      if (outPlain) {
        outPlain[(size_t)m * N + n] = val;
      } else {
        int hh = n / headDim, dd = n % headDim;
        int b = m / L_, l = m % L_;
        size_t idx = vTrans
            ? (((size_t)b * H_ + hh) * headDim + dd) * L_ + l   // [B,H,d,L]
            : (((size_t)b * H_ + hh) * L_ + l) * headDim + dd;  // [B,H,L,d]
        outSplit[idx] = f2bf(val);
      }
    }
}

// ---------- fused relative-position flash attention --------------------------
// grid.x = B*H, grid.y = L/64; block = 128 threads (4 waves, 16 rows each).
// K/V tiles are streamed into double-buffered LDS with async global->LDS copies
// (ASYNCcnt) so tile t+1 transfers overlap tile t compute.
__global__ __launch_bounds__(128) void attn_fused(
    const __bf16* __restrict__ Qp,   // [B,H,L,32]
    const __bf16* __restrict__ Kp,   // [B,H,L,32]
    const __bf16* __restrict__ Vt,   // [B,H,64,L]  (transposed V)
    const __bf16* __restrict__ Eb,   // [EPAD_,32] (zero padded)
    __bf16* __restrict__ Out)        // [B,L,256] bf16
{
  __shared__ __align__(16) __bf16 kTile[2][64][DQK_]; // 2 x 4KB
  __shared__ __align__(16) __bf16 vTile[2][64][64];   // 2 x 8KB
  __shared__ float  qeLds[4][16][QESTR];
  __shared__ __align__(16) __bf16 pLds[4][16][PSTR];

  const int tid  = threadIdx.x;
  const int bh   = blockIdx.x;
  const int i0   = blockIdx.y * 64;
  const int wave = tid >> 5;
  const int lane = tid & 31;
  const int ln   = lane & 15;
  const int half = lane >> 4;
  const int iw   = i0 + wave * 16;       // first row of this wave's strip

  const __bf16* Qb = Qp + ((size_t)bh * L_ + iw) * DQK_;
  const __bf16* Kb = Kp + (size_t)bh * L_ * DQK_;
  const __bf16* Vb = Vt + (size_t)bh * DH_ * L_;

  // Q A-fragment (persistent: K=32 == DQK)
  v16bf qa = load_a_frag(Qb + (size_t)ln * DQK_, half);

  v8f   o[4] = {};
  float mrun[8], lrun[8];
#pragma unroll
  for (int v = 0; v < 8; ++v) { mrun[v] = -3.0e38f; lrun[v] = 0.0f; }

  // stage one K/V tile into LDS buffer `buf` (cooperative, per-lane 16B copies)
  auto stage = [&](int j0, int buf) {
    // K tile: 64 rows x 32 bf16 = contiguous 4KB; 32B per thread
    const char* gk = (const char*)(Kb + (size_t)j0 * DQK_);
    unsigned    lk = (unsigned)(size_t)&kTile[buf][0][0];
    async_copy_b128(lk + tid * 32,      gk + (size_t)tid * 32);
    async_copy_b128(lk + tid * 32 + 16, gk + (size_t)tid * 32 + 16);
    // V tile: 64 rows (head dim) x 64 bf16, row stride L_; 64B per thread
    int d = tid >> 1, jh = tid & 1;
    const char* gv = (const char*)(Vb + (size_t)d * L_ + j0 + jh * 32);
    unsigned    lv = (unsigned)(size_t)&vTile[buf][d][jh * 32];
#pragma unroll
    for (int c = 0; c < 4; ++c)
      async_copy_b128(lv + c * 16, gv + c * 16);
  };

  const int ntiles = i0 / 64 + 1;        // causal: cols 0 .. i0+63
  stage(0, 0);                           // prologue

  for (int t = 0; t < ntiles; ++t) {
    const int j0 = t * 64;
    wait_async0();                       // tile t resident in LDS
    __syncthreads();                     // visible to all waves
    if (t + 1 < ntiles)                  // stream tile t+1 behind compute
      stage(j0 + 64, (t + 1) & 1);

    const __bf16 (*kT)[DQK_] = kTile[t & 1];
    const __bf16 (*vT)[64]   = vTile[t & 1];

    // ---- QE band: QE[i, m], m in [mbase, mbase+96) ; 6 WMMAs -> LDS
    const int mbase = L_ - 16 - iw + j0;
    const __bf16* Erow0 = Eb + (size_t)(mbase + ln) * DQK_;
    v16bf ef[6];
#pragma unroll
    for (int ct = 0; ct < 6; ++ct)
      ef[ct] = load_b_frag(Erow0 + (size_t)ct * 16 * DQK_, half);
#pragma unroll
    for (int ct = 0; ct < 6; ++ct) {
      v8f qe = {};
      qe = __builtin_amdgcn_wmma_f32_16x16x32_bf16(
          false, qa, false, ef[ct], (short)0, qe, false, false);
#pragma unroll
      for (int v = 0; v < 8; ++v)
        qeLds[wave][v + 8 * half][ct * 16 + ln] = qe[v];
    }

    // ---- Q*K^T : 4 WMMAs; K-frags from LDS (shared by all 4 waves)
    v16bf kf[4];
#pragma unroll
    for (int nt = 0; nt < 4; ++nt)
      kf[nt] = load_b_frag(&kT[nt * 16 + ln][0], half);
    v8f s[4];
#pragma unroll
    for (int nt = 0; nt < 4; ++nt) {
      v8f z = {};
      s[nt] = __builtin_amdgcn_wmma_f32_16x16x32_bf16(
          false, qa, false, kf[nt], (short)0, z, false, false);
    }

    // ---- skew gather + causal mask
#pragma unroll
    for (int nt = 0; nt < 4; ++nt)
#pragma unroll
      for (int v = 0; v < 8; ++v) {
        int rv = v + 8 * half;
        int i  = iw + rv;
        int cj = nt * 16 + ln;
        int j  = j0 + cj;
        float srel = qeLds[wave][rv][15 - rv + cj];  // QE[i, L-1-(i-j)]
        s[nt][v] = (j <= i) ? (s[nt][v] + srel) : -1.0e9f;
      }

    // ---- online softmax (row reductions across 16 lanes of each half)
#pragma unroll
    for (int v = 0; v < 8; ++v) {
      float mx = fmaxf(fmaxf(s[0][v], s[1][v]), fmaxf(s[2][v], s[3][v]));
#pragma unroll
      for (int off = 1; off < 16; off <<= 1)
        mx = fmaxf(mx, __shfl_xor(mx, off, 32));
      float mnew  = fmaxf(mrun[v], mx);
      float scale = __expf(mrun[v] - mnew);
      mrun[v] = mnew;
      float rs = 0.0f;
#pragma unroll
      for (int nt = 0; nt < 4; ++nt) {
        float p = __expf(s[nt][v] - mnew);
        s[nt][v] = p;
        rs += p;
      }
#pragma unroll
      for (int off = 1; off < 16; off <<= 1)
        rs += __shfl_xor(rs, off, 32);
      lrun[v] = lrun[v] * scale + rs;
#pragma unroll
      for (int nt = 0; nt < 4; ++nt) o[nt][v] *= scale;
    }

    // ---- P -> LDS (bf16) so we can reload it in A-fragment layout
#pragma unroll
    for (int nt = 0; nt < 4; ++nt)
#pragma unroll
      for (int v = 0; v < 8; ++v)
        pLds[wave][v + 8 * half][nt * 16 + ln] = f2bf(s[nt][v]);

    // ---- O += P(16x64) @ V(64x64) : 2 K-chunks x 4 subtiles = 8 WMMAs
#pragma unroll
    for (int ck = 0; ck < 2; ++ck) {
      v16bf pa = load_a_frag(&pLds[wave][ln][ck * 32], half);
      v16bf vf[4];
#pragma unroll
      for (int nt = 0; nt < 4; ++nt)
        vf[nt] = load_b_frag(&vT[nt * 16 + ln][ck * 32], half);
#pragma unroll
      for (int nt = 0; nt < 4; ++nt)
        o[nt] = __builtin_amdgcn_wmma_f32_16x16x32_bf16(
            false, pa, false, vf[nt], (short)0, o[nt], false, false);
    }
  }

  // ---- epilogue: softmax normalize, post-softmax /sqrt(64), head-concat store
  const int b = bh / H_, h = bh % H_;
#pragma unroll
  for (int nt = 0; nt < 4; ++nt)
#pragma unroll
    for (int v = 0; v < 8; ++v) {
      int i = iw + v + 8 * half;
      int d = nt * 16 + ln;
      float val = o[nt][v] / (lrun[v] * 8.0f);
      Out[((size_t)b * L_ + i) * D_ + h * DH_ + d] = f2bf(val);
    }
}

// ---------- host launch ------------------------------------------------------
extern "C" void kernel_launch(void* const* d_in, const int* in_sizes, int n_in,
                              void* d_out, int out_size, void* d_ws, size_t ws_size,
                              hipStream_t stream) {
  const float* q_in = (const float*)d_in[0];
  const float* k_in = (const float*)d_in[1];
  const float* v_in = (const float*)d_in[2];
  const float* Wq   = (const float*)d_in[3];
  const float* bq   = (const float*)d_in[4];
  const float* Wk   = (const float*)d_in[5];
  const float* bk   = (const float*)d_in[6];
  const float* Wv   = (const float*)d_in[7];
  const float* bv   = (const float*)d_in[8];
  const float* Wfc  = (const float*)d_in[9];
  const float* bfc  = (const float*)d_in[10];
  const float* E    = (const float*)d_in[11];
  float* out        = (float*)d_out;

  const size_t nBLD = (size_t)B_ * L_ * D_;          // 2,097,152
  size_t off = 0;
  auto carve = [&](size_t bytes) {
    char* p = (char*)d_ws + off;
    off += (bytes + 255) & ~(size_t)255;
    return p;
  };
  __bf16* qbf   = (__bf16*)carve(nBLD * 2);
  __bf16* kbf   = (__bf16*)carve(nBLD * 2);
  __bf16* vbf   = (__bf16*)carve(nBLD * 2);
  __bf16* wqbf  = (__bf16*)carve((size_t)D_ * (D_ / 2) * 2);   // transposed [128,256]
  __bf16* wkbf  = (__bf16*)carve((size_t)D_ * (D_ / 2) * 2);   // transposed [128,256]
  __bf16* wvbf  = (__bf16*)carve((size_t)D_ * D_ * 2);         // transposed [256,256]
  __bf16* wfcbf = (__bf16*)carve((size_t)D_ * D_ * 2);         // transposed [256,256]
  __bf16* ebf   = (__bf16*)carve((size_t)EPAD_ * DQK_ * 2);
  __bf16* qp    = (__bf16*)carve((size_t)B_ * H_ * L_ * DQK_ * 2);
  __bf16* kp    = (__bf16*)carve((size_t)B_ * H_ * L_ * DQK_ * 2);
  __bf16* vpt   = (__bf16*)carve((size_t)B_ * H_ * DH_ * L_ * 2);  // transposed V
  __bf16* abf   = (__bf16*)carve(nBLD * 2);          // attention output, bf16

  auto cvt = [&](const float* s, __bf16* d, int n) {
    cvt_f32_bf16<<<(n + 255) / 256, 256, 0, stream>>>(s, d, n);
  };
  auto cvtT = [&](const float* s, __bf16* d, int K, int N) {
    cvt_f32_bf16_T<<<(K * N + 255) / 256, 256, 0, stream>>>(s, d, K, N);
  };
  cvt(q_in, qbf, (int)nBLD);
  cvt(k_in, kbf, (int)nBLD);
  cvt(v_in, vbf, (int)nBLD);
  cvtT(Wq, wqbf, D_, D_ / 2);
  cvtT(Wk, wkbf, D_, D_ / 2);
  cvtT(Wv, wvbf, D_, D_);
  cvtT(Wfc, wfcbf, D_, D_);
  cvt(E, ebf, L_ * DQK_);
  zero_bf16<<<(128 * DQK_ + 255) / 256, 256, 0, stream>>>(ebf + (size_t)L_ * DQK_, 128 * DQK_);

  const int M = B_ * L_;                              // 8192
  // projections
  gemm_bf16<<<dim3(M / 64, (D_ / 2) / 64), 128, 0, stream>>>(
      qbf, wqbf, bq, qp, nullptr, D_ / 2, DQK_, 0);
  gemm_bf16<<<dim3(M / 64, (D_ / 2) / 64), 128, 0, stream>>>(
      kbf, wkbf, bk, kp, nullptr, D_ / 2, DQK_, 0);
  gemm_bf16<<<dim3(M / 64, D_ / 64), 128, 0, stream>>>(
      vbf, wvbf, bv, vpt, nullptr, D_, DH_, 1);       // V stored transposed
  // fused attention
  attn_fused<<<dim3(B_ * H_, L_ / 64), 128, 0, stream>>>(qp, kp, vpt, ebf, abf);
  // final FC -> fp32 output
  gemm_bf16<<<dim3(M / 64, D_ / 64), 128, 0, stream>>>(
      abf, wfcbf, bfc, nullptr, out, D_, DH_, 0);
}
